// MS3DV_472446402672
// MI455X (gfx1250) — compile-verified, matrix operand order checked by reading, not run on hardware
//
#include <hip/hip_runtime.h>
#include <hip/hip_bf16.h>

// ---------------------------------------------------------------------------
// CDNA5 (gfx1250) implementation of the multi-scale 3D volume network.
// All channel contractions (2D down-convs, 3D convs, FC layers) are implicit
// GEMMs on v_wmma_f32_16x16x32_f16 (f16 A/B, f32 accumulate).
// K is re-ordered as tap*64+ic so per-element staging math is shift/mask only.
// ---------------------------------------------------------------------------

typedef __attribute__((ext_vector_type(16))) _Float16 v16h;
typedef __attribute__((ext_vector_type(8)))  _Float16 v8h;
typedef __attribute__((ext_vector_type(8)))  float    v8f;

#define WMMA_F16(a, b, c) \
  __builtin_amdgcn_wmma_f32_16x16x32_f16(false, (a), false, (b), (short)0, (c), false, false)

// --- A operand: 16x32 f16 tile in LDS, per-lane load per ISA layout ---------
// lane<16 : M=lane,    K halves {0..7} then {16..23}
// lane>=16: M=lane-16, K halves {8..15} then {24..31}
__device__ __forceinline__ v16h load_A_lds(const _Float16* tile, int lane,
                                           int rowStrideHalves, int kofs) {
  int m   = lane & 15;
  int kh0 = kofs + ((lane & 16) ? 8 : 0);
  const v8h* p0 = (const v8h*)(tile + m * rowStrideHalves + kh0);
  const v8h* p1 = (const v8h*)(tile + m * rowStrideHalves + kh0 + 16);
  v8h a0 = *p0, a1 = *p1;
  v16h a;
#pragma unroll
  for (int j = 0; j < 8; ++j) { a[j] = a0[j]; a[8 + j] = a1[j]; }
  return a;
}

// --- B operand: pre-packed per-lane (32B contiguous per lane, coalesced) ----
__device__ __forceinline__ v16h load_B_pack(const _Float16* packB, long halfIdx) {
  const v8h* p = (const v8h*)(packB + halfIdx);
  v8h b0 = p[0], b1 = p[1];
  v16h b;
#pragma unroll
  for (int j = 0; j < 8; ++j) { b[j] = b0[j]; b[8 + j] = b1[j]; }
  return b;
}

// ---------------------------------------------------------------------------
// Weight packing -> WMMA B-operand striping (f16).
// dst flat index = ((ocBlk*nK + s)*32 + lane)*16 + j.
// GEMM K index: taps==0 -> k directly (FC);  taps>0 -> k = tap*64 + ic with
// source w[oc][ic][tap] = w[oc*K + ic*taps + tap] (conv OIHW/OIDHW).
// ---------------------------------------------------------------------------
__global__ void kcdna5_pack_b(const float* __restrict__ w, _Float16* __restrict__ dst,
                              int OC, int K, int taps) {
  int nK = K >> 5;
  long total = (long)OC * K;
  for (long i = (long)blockIdx.x * blockDim.x + threadIdx.x; i < total;
       i += (long)gridDim.x * blockDim.x) {
    int j    = (int)(i & 15);
    int lane = (int)((i >> 4) & 31);
    long g   = i >> 9;                 // ocBlk*nK + s
    int s    = (int)(g % nK);
    int ocb  = (int)(g / nK);
    int oc   = ocb * 16 + (lane & 15);
    int kl   = (j & 7) + ((lane & 16) ? 8 : 0) + ((j & 8) ? 16 : 0);
    int k    = s * 32 + kl;
    long srcIdx;
    if (taps == 0) srcIdx = (long)oc * K + k;
    else           srcIdx = (long)oc * K + (long)(k & 63) * taps + (k >> 6);
    dst[i] = (_Float16)w[srcIdx];
  }
}

// ---------------------------------------------------------------------------
// 2D conv, 3x3, stride 2, pad 1, 64->64 ch. Implicit GEMM, K = 9*64 = 576.
// Block: 128 threads = 4 waves; 16 output pixels x 64 out-channels.
// Ws = 1<<lw (power of two). Each thread stages k = lane for m = wv + 4u.
// ---------------------------------------------------------------------------
__global__ void kcdna5_conv2d_s2(const float* __restrict__ src,
                                 const _Float16* __restrict__ packB,
                                 const float* __restrict__ bias,
                                 float* __restrict__ dst,
                                 int Hin, int Win, int lw, int HsWs) {
  __shared__ _Float16 As[16 * 40];
  const int nK = 18;                              // 576 / 32
  int m_base = blockIdx.x * 16;
  int img    = blockIdx.y;
  int tid  = threadIdx.x;
  int lane = tid & 31;
  int wv   = tid >> 5;                            // oc block 0..3
  int Wm1  = (1 << lw) - 1;
  const float* base = src + (long)img * 64 * Hin * Win;
  // per-thread staged rows: m = wv + 4u, fixed k = lane
  int oy[4], ox[4];
#pragma unroll
  for (int u = 0; u < 4; ++u) {
    int pix = m_base + wv + 4 * u;
    oy[u] = pix >> lw;
    ox[u] = pix & Wm1;
  }
  v8f acc = {};
  for (int s = 0; s < nK; ++s) {
    int t  = s >> 1;                              // tap (uniform -> SALU)
    int dy = t / 3 - 1;
    int dx = t - (t / 3) * 3 - 1;
    int ic = lane + ((s & 1) << 5);
    const float* cbase = base + (long)ic * Hin * Win;
    __syncthreads();
#pragma unroll
    for (int u = 0; u < 4; ++u) {
      int iy = 2 * oy[u] + dy;
      int ix = 2 * ox[u] + dx;
      float v = 0.f;
      if ((unsigned)iy < (unsigned)Hin && (unsigned)ix < (unsigned)Win)
        v = cbase[(long)iy * Win + ix];
      As[(wv + 4 * u) * 40 + lane] = (_Float16)v;
    }
    __syncthreads();
    if (s + 1 < nK)
      __builtin_prefetch((const void*)(packB + ((long)(wv * nK + s + 1) * 32 + lane) * 16), 0, 3);
    v16h a = load_A_lds(As, lane, 40, 0);
    v16h b = load_B_pack(packB, ((long)(wv * nK + s) * 32 + lane) * 16);
    acc = WMMA_F16(a, b, acc);
  }
  int oc = wv * 16 + (lane & 15);
  float bo = bias[oc];
  int Mo = (lane & 16) ? 8 : 0;
#pragma unroll
  for (int r = 0; r < 8; ++r) {
    int pix = m_base + r + Mo;
    dst[((long)(img * 64 + oc)) * HsWs + pix] = acc[r] + bo;
  }
}

// ---------------------------------------------------------------------------
// 3D conv, 64->64 ch, taps = 1 (1x1x1, pad 0) or 27 (3x3x3, pad 1).
// res = 1<<lr. Block: 128 threads = 4 waves; 16 voxels x 64 out-channels.
// ---------------------------------------------------------------------------
__global__ void kcdna5_conv3d(const float* __restrict__ src,
                              const _Float16* __restrict__ packB,
                              const float* __restrict__ bias,
                              float* __restrict__ dst,
                              int lr, int taps) {
  __shared__ _Float16 As[16 * 40];
  int res  = 1 << lr;
  int res3 = 1 << (3 * lr);
  int rm1  = res - 1;
  int nK = (64 * taps) >> 5;
  int m_base = blockIdx.x * 16;
  int b  = blockIdx.y;
  int tid = threadIdx.x, lane = tid & 31, wv = tid >> 5;
  const float* base = src + (long)b * 64 * res3;
  int vz[4], vy[4], vx[4];
#pragma unroll
  for (int u = 0; u < 4; ++u) {
    int vox = m_base + wv + 4 * u;
    vz[u] = vox >> (2 * lr);
    vy[u] = (vox >> lr) & rm1;
    vx[u] = vox & rm1;
  }
  v8f acc = {};
  for (int s = 0; s < nK; ++s) {
    int t, icbase;
    if (taps == 1) { t = 0; icbase = s << 5; }
    else           { t = s >> 1; icbase = (s & 1) << 5; }
    int t9 = t / 9;
    int t2 = t - t9 * 9;
    int dz = t9 - 1, dy = t2 / 3 - 1, dx = t2 - (t2 / 3) * 3 - 1;
    if (taps == 1) { dz = dy = dx = 0; }
    int ic = lane + icbase;
    const float* cbase = base + (long)ic * res3;
    __syncthreads();
#pragma unroll
    for (int u = 0; u < 4; ++u) {
      int z = vz[u] + dz, y = vy[u] + dy, x = vx[u] + dx;
      float v = 0.f;
      if ((unsigned)z < (unsigned)res && (unsigned)y < (unsigned)res &&
          (unsigned)x < (unsigned)res)
        v = cbase[(((z << lr) + y) << lr) + x];
      As[(wv + 4 * u) * 40 + lane] = (_Float16)v;
    }
    __syncthreads();
    if (s + 1 < nK)
      __builtin_prefetch((const void*)(packB + ((long)(wv * nK + s + 1) * 32 + lane) * 16), 0, 3);
    v16h a  = load_A_lds(As, lane, 40, 0);
    v16h bb = load_B_pack(packB, ((long)(wv * nK + s) * 32 + lane) * 16);
    acc = WMMA_F16(a, bb, acc);
  }
  int oc = wv * 16 + (lane & 15);
  float bo = bias[oc];
  int Mo = (lane & 16) ? 8 : 0;
#pragma unroll
  for (int r = 0; r < 8; ++r)
    dst[((long)(b * 64 + oc)) * res3 + m_base + r + Mo] = acc[r] + bo;
}

// ---------------------------------------------------------------------------
// InstanceNorm over one (image,channel) plane + optional residual + LeakyReLU.
// out = lrelu(norm(x) [+ resid], slope).  One block (256 thr) per plane.
// ---------------------------------------------------------------------------
__global__ void kcdna5_inorm(const float* x, const float* resid, float* out,
                             int plane, float slope) {
  __shared__ float s_sum[256];
  __shared__ float s_sq[256];
  long p = blockIdx.x;
  const float* xi = x + p * plane;
  float sum = 0.f, sq = 0.f;
  for (int i = threadIdx.x; i < plane; i += 256) {
    float v = xi[i];
    sum += v; sq += v * v;
  }
  s_sum[threadIdx.x] = sum; s_sq[threadIdx.x] = sq;
  __syncthreads();
  for (int o = 128; o > 0; o >>= 1) {
    if ((int)threadIdx.x < o) {
      s_sum[threadIdx.x] += s_sum[threadIdx.x + o];
      s_sq[threadIdx.x]  += s_sq[threadIdx.x + o];
    }
    __syncthreads();
  }
  float m   = s_sum[0] / (float)plane;
  float var = s_sq[0] / (float)plane - m * m;
  float inv = rsqrtf(var + 1e-5f);
  const float* ri = resid ? resid + p * plane : nullptr;
  float* oi = out + p * plane;
  for (int i = threadIdx.x; i < plane; i += 256) {
    float v = (xi[i] - m) * inv;
    if (ri) v += ri[i];
    oi[i] = (v >= 0.f) ? v : slope * v;
  }
}

// ---------------------------------------------------------------------------
// Build 3D volume: bilinear sample 3 view planes (align_corners), max-reduce.
// pf layout [(b*3+v)*64 + c][Hs][Hs]; vol layout [(b*64+c)][res^3]; res=1<<lr.
// ---------------------------------------------------------------------------
__global__ void kcdna5_buildvol(const float* __restrict__ pf, float* __restrict__ vol,
                                int Hs, int lr, int B) {
  int res = 1 << lr;
  int rm1 = res - 1;
  long res3 = (long)1 << (3 * lr);
  long total = (long)B * 64 * res3;
  float scale = (float)(Hs - 1) / (float)(res - 1);
  for (long idx = (long)blockIdx.x * 256 + threadIdx.x; idx < total;
       idx += (long)gridDim.x * 256) {
    int m = (int)(idx & (res3 - 1));
    int c = (int)(idx >> (3 * lr)) & 63;
    int b = (int)(idx >> (3 * lr + 6));
    int a  = m >> (2 * lr);
    int r1 = (m >> lr) & rm1;
    int r2 = m & rm1;
    int av[3] = {a, a, r1};
    int bv[3] = {r1, r2, r2};
    float best = -1e30f;
#pragma unroll
    for (int v = 0; v < 3; ++v) {
      float fi = av[v] * scale, fj = bv[v] * scale;
      int i0 = (int)fi; if (i0 > Hs - 1) i0 = Hs - 1;
      int j0 = (int)fj; if (j0 > Hs - 1) j0 = Hs - 1;
      int i1 = i0 + 1 < Hs ? i0 + 1 : Hs - 1;
      int j1 = j0 + 1 < Hs ? j0 + 1 : Hs - 1;
      float wi = fi - (float)i0, wj = fj - (float)j0;
      const float* P = pf + ((long)((b * 3 + v) * 64 + c)) * Hs * Hs;
      float g00 = P[i0 * Hs + j0], g01 = P[i0 * Hs + j1];
      float g10 = P[i1 * Hs + j0], g11 = P[i1 * Hs + j1];
      float val = (g00 * (1.f - wj) + g01 * wj) * (1.f - wi) +
                  (g10 * (1.f - wj) + g11 * wj) * wi;
      best = fmaxf(best, val);
    }
    vol[idx] = best;
  }
}

// ---------------------------------------------------------------------------
// Fused: trilinear sample 3 volumes -> cat[16][192] f16 in LDS ->
//        FC1 (K=192, WMMA) + ReLU -> LDS -> FC2 (K=128, WMMA) -> out f32.
// Block: 256 threads = 8 waves; 16 query points per block. N = 2^17.
// ---------------------------------------------------------------------------
#define MLP_LOGN 17
__global__ void kcdna5_mlp(const float* __restrict__ q,
                           const float* __restrict__ v0,
                           const float* __restrict__ v1,
                           const float* __restrict__ v2,
                           const _Float16* __restrict__ packFc1,
                           const float* __restrict__ fc1b,
                           const _Float16* __restrict__ packFc2,
                           const float* __restrict__ fc2b,
                           float* __restrict__ out) {
  __shared__ _Float16 Cf[16 * 200];   // 16 points x 192 cat features (padded)
  __shared__ _Float16 Hf[16 * 136];   // 16 points x 128 hidden (padded)
  int p_base = blockIdx.x * 16;
  int tid = threadIdx.x;
  int lane = tid & 31;
  int wv = tid >> 5;                   // 0..7 -> 16-wide output-channel block
  // ---- gather: trilinear sampling (align_corners) from the three volumes ----
  for (int f = tid; f < 16 * 192; f += 256) {
    int pt = f / 192, ch = f - pt * 192;
    int sc = ch >> 6, c = ch & 63;
    int p = p_base + pt;
    int b = p >> MLP_LOGN;
    const float* qp = q + (long)p * 3;
    float qx = qp[0], qy = qp[1], qz = qp[2];
    const float* vol = (sc == 0) ? v0 : ((sc == 1) ? v1 : v2);
    int lr = 5 - sc;
    int res = 1 << lr;
    long res3 = (long)1 << (3 * lr);
    float h = 0.5f * (float)(res - 1);
    float fx = (qx + 1.f) * h, fy = (qy + 1.f) * h, fz = (qz + 1.f) * h;
    fx = fminf(fmaxf(fx, 0.f), (float)(res - 1));
    fy = fminf(fmaxf(fy, 0.f), (float)(res - 1));
    fz = fminf(fmaxf(fz, 0.f), (float)(res - 1));
    int x0 = (int)fx; if (x0 > res - 1) x0 = res - 1;
    int y0 = (int)fy; if (y0 > res - 1) y0 = res - 1;
    int z0 = (int)fz; if (z0 > res - 1) z0 = res - 1;
    int x1 = x0 + 1 < res ? x0 + 1 : res - 1;
    int y1 = y0 + 1 < res ? y0 + 1 : res - 1;
    int z1 = z0 + 1 < res ? z0 + 1 : res - 1;
    float wx = fx - (float)x0, wy = fy - (float)y0, wz = fz - (float)z0;
    const float* base = vol + ((long)(b * 64 + c)) * res3;
    float c000 = base[(((z0 << lr) + y0) << lr) + x0];
    float c001 = base[(((z0 << lr) + y0) << lr) + x1];
    float c010 = base[(((z0 << lr) + y1) << lr) + x0];
    float c011 = base[(((z0 << lr) + y1) << lr) + x1];
    float c100 = base[(((z1 << lr) + y0) << lr) + x0];
    float c101 = base[(((z1 << lr) + y0) << lr) + x1];
    float c110 = base[(((z1 << lr) + y1) << lr) + x0];
    float c111 = base[(((z1 << lr) + y1) << lr) + x1];
    float c00 = c000 * (1.f - wx) + c001 * wx;
    float c01 = c010 * (1.f - wx) + c011 * wx;
    float c10 = c100 * (1.f - wx) + c101 * wx;
    float c11 = c110 * (1.f - wx) + c111 * wx;
    float c0 = c00 * (1.f - wy) + c01 * wy;
    float c1 = c10 * (1.f - wy) + c11 * wy;
    Cf[pt * 200 + ch] = (_Float16)(c0 * (1.f - wz) + c1 * wz);
  }
  __syncthreads();
  // ---- FC1: M=16 pts, N=16 (per wave), K=192 -> 6 WMMA steps ----
  v8f acc = {};
#pragma unroll
  for (int s = 0; s < 6; ++s) {
    v16h a = load_A_lds(Cf, lane, 200, s * 32);
    v16h b = load_B_pack(packFc1, ((long)(wv * 6 + s) * 32 + lane) * 16);
    acc = WMMA_F16(a, b, acc);
  }
  int oc = wv * 16 + (lane & 15);
  int Mo = (lane & 16) ? 8 : 0;
  float bo = fc1b[oc];
#pragma unroll
  for (int r = 0; r < 8; ++r) {
    float hv = acc[r] + bo;
    hv = hv > 0.f ? hv : 0.f;
    Hf[(r + Mo) * 136 + oc] = (_Float16)hv;
  }
  __syncthreads();
  // ---- FC2: M=16 pts, N=16 (per wave), K=128 -> 4 WMMA steps ----
  v8f acc2 = {};
#pragma unroll
  for (int s = 0; s < 4; ++s) {
    v16h a = load_A_lds(Hf, lane, 136, s * 32);
    v16h b = load_B_pack(packFc2, ((long)(wv * 4 + s) * 32 + lane) * 16);
    acc2 = WMMA_F16(a, b, acc2);
  }
  float bo2 = fc2b[oc];
#pragma unroll
  for (int r = 0; r < 8; ++r)
    out[((long)(p_base + r + Mo)) * 128 + oc] = acc2[r] + bo2;
}

// ---------------------------------------------------------------------------
// Host orchestration.
// Input flattening assumption (dict order, params as sorted-key pytree):
//  0 proj_feats [2,3,64,256,256]   1 query_points [2,131072,3]
//  2..31  c3d (3 scales x {proj_w, proj_b, res0{b1,b2,w1,w2}, res1{b1,b2,w1,w2}})
//  32..37 down: d1.w d1.b d2a.w d2a.b d2b.w d2b.b
//  38 fc1_b  39 fc1_w  40 fc2_b  41 fc2_w
// ---------------------------------------------------------------------------
extern "C" void kernel_launch(void* const* d_in, const int* in_sizes, int n_in,
                              void* d_out, int out_size, void* d_ws, size_t ws_size,
                              hipStream_t stream) {
  const int B = 2, N = 131072;
  const int RES[3] = {32, 16, 8};
  const int LR[3]  = {5, 4, 3};

  const float* proj = (const float*)d_in[0];
  const float* qpts = (const float*)d_in[1];
  const float *c3d_pw[3], *c3d_pb[3];
  const float *rb_b1[3][2], *rb_b2[3][2], *rb_w1[3][2], *rb_w2[3][2];
  for (int i = 0; i < 3; ++i) {
    int base = 2 + i * 10;
    c3d_pw[i] = (const float*)d_in[base + 0];
    c3d_pb[i] = (const float*)d_in[base + 1];
    for (int r = 0; r < 2; ++r) {
      rb_b1[i][r] = (const float*)d_in[base + 2 + r * 4];
      rb_b2[i][r] = (const float*)d_in[base + 3 + r * 4];
      rb_w1[i][r] = (const float*)d_in[base + 4 + r * 4];
      rb_w2[i][r] = (const float*)d_in[base + 5 + r * 4];
    }
  }
  const float* dw1  = (const float*)d_in[32];
  const float* db1  = (const float*)d_in[33];
  const float* dw2a = (const float*)d_in[34];
  const float* db2a = (const float*)d_in[35];
  const float* dw2b = (const float*)d_in[36];
  const float* db2b = (const float*)d_in[37];
  const float* fc1b = (const float*)d_in[38];
  const float* fc1w = (const float*)d_in[39];
  const float* fc2b = (const float*)d_in[40];
  const float* fc2w = (const float*)d_in[41];
  float* out = (float*)d_out;

  // ---- workspace carving ----
  char* W = (char*)d_ws;
  size_t off = 0;
  auto carve = [&](size_t bytes) -> void* {
    void* p = W + off;
    off += (bytes + 255) & ~(size_t)255;
    return p;
  };
  _Float16* pk_d1  = (_Float16*)carve((size_t)64 * 576 * 2);
  _Float16* pk_d2a = (_Float16*)carve((size_t)64 * 576 * 2);
  _Float16* pk_d2b = (_Float16*)carve((size_t)64 * 576 * 2);
  _Float16* pk_proj[3];
  _Float16* pk_rb[3][2][2];
  for (int i = 0; i < 3; ++i) {
    pk_proj[i] = (_Float16*)carve((size_t)64 * 64 * 2);
    for (int r = 0; r < 2; ++r)
      for (int j = 0; j < 2; ++j)
        pk_rb[i][r][j] = (_Float16*)carve((size_t)64 * 1728 * 2);
  }
  _Float16* pk_fc1 = (_Float16*)carve((size_t)128 * 192 * 2);
  _Float16* pk_fc2 = (_Float16*)carve((size_t)128 * 128 * 2);
  float* t2d_big   = (float*)carve((size_t)6 * 64 * 128 * 128 * 4);
  float* t2d_small = (float*)carve((size_t)6 * 64 * 64 * 64 * 4);
  float *volA[3], *volB[3], *t1[3], *t2[3];
  for (int i = 0; i < 3; ++i) {
    size_t vb = (size_t)B * 64 * RES[i] * RES[i] * RES[i] * 4;
    volA[i] = (float*)carve(vb);
    volB[i] = (float*)carve(vb);
    t1[i]   = (float*)carve(vb);
    t2[i]   = (float*)carve(vb);
  }

  // ---- 1. pack all weights into WMMA-B layout (f16) ----
  auto pack = [&](const float* src, _Float16* dst, int OC, int K, int taps) {
    int total = OC * K;
    kcdna5_pack_b<<<dim3((total + 255) / 256), dim3(256), 0, stream>>>(src, dst, OC, K, taps);
  };
  pack(dw1, pk_d1, 64, 576, 9);
  pack(dw2a, pk_d2a, 64, 576, 9);
  pack(dw2b, pk_d2b, 64, 576, 9);
  for (int i = 0; i < 3; ++i) {
    pack(c3d_pw[i], pk_proj[i], 64, 64, 1);
    for (int r = 0; r < 2; ++r) {
      pack(rb_w1[i][r], pk_rb[i][r][0], 64, 1728, 27);
      pack(rb_w2[i][r], pk_rb[i][r][1], 64, 1728, 27);
    }
  }
  pack(fc1w, pk_fc1, 128, 192, 0);
  pack(fc2w, pk_fc2, 128, 128, 0);

  // ---- 2. scale 1: volume straight from proj_feats (Hs = 256, res = 32) ----
  {
    long total = (long)B * 64 * 32 * 32 * 32;
    kcdna5_buildvol<<<dim3((unsigned)((total + 255) / 256)), dim3(256), 0, stream>>>(
        proj, volA[0], 256, 5, B);
  }
  // ---- 3. scale 2: conv2d s2 -> inorm+lrelu(0.01) -> volume (res 16) ----
  kcdna5_conv2d_s2<<<dim3(128 * 128 / 16, 6), dim3(128), 0, stream>>>(
      proj, pk_d1, db1, t2d_big, 256, 256, 7, 128 * 128);
  kcdna5_inorm<<<dim3(6 * 64), dim3(256), 0, stream>>>(
      t2d_big, nullptr, t2d_big, 128 * 128, 0.01f);
  {
    long total = (long)B * 64 * 16 * 16 * 16;
    kcdna5_buildvol<<<dim3((unsigned)((total + 255) / 256)), dim3(256), 0, stream>>>(
        t2d_big, volA[1], 128, 4, B);
  }
  // ---- 4. scale 4: two conv2d s2 stages -> volume (res 8) ----
  kcdna5_conv2d_s2<<<dim3(128 * 128 / 16, 6), dim3(128), 0, stream>>>(
      proj, pk_d2a, db2a, t2d_big, 256, 256, 7, 128 * 128);
  kcdna5_inorm<<<dim3(6 * 64), dim3(256), 0, stream>>>(
      t2d_big, nullptr, t2d_big, 128 * 128, 0.01f);
  kcdna5_conv2d_s2<<<dim3(64 * 64 / 16, 6), dim3(128), 0, stream>>>(
      t2d_big, pk_d2b, db2b, t2d_small, 128, 128, 6, 64 * 64);
  kcdna5_inorm<<<dim3(6 * 64), dim3(256), 0, stream>>>(
      t2d_small, nullptr, t2d_small, 64 * 64, 0.01f);
  {
    long total = (long)B * 64 * 8 * 8 * 8;
    kcdna5_buildvol<<<dim3((unsigned)((total + 255) / 256)), dim3(256), 0, stream>>>(
        t2d_small, volA[2], 64, 3, B);
  }

  // ---- 5. per-scale 3D pipeline: 1x1 proj conv + 2 ResBlocks ----
  for (int i = 0; i < 3; ++i) {
    int res = RES[i];
    int res3 = res * res * res;
    dim3 cg(res3 / 16, B);
    kcdna5_conv3d<<<cg, dim3(128), 0, stream>>>(volA[i], pk_proj[i], c3d_pb[i],
                                                volB[i], LR[i], 1);
    for (int r = 0; r < 2; ++r) {
      kcdna5_conv3d<<<cg, dim3(128), 0, stream>>>(volB[i], pk_rb[i][r][0],
                                                  rb_b1[i][r], t1[i], LR[i], 27);
      kcdna5_inorm<<<dim3(B * 64), dim3(256), 0, stream>>>(
          t1[i], nullptr, t1[i], res3, 0.2f);
      kcdna5_conv3d<<<cg, dim3(128), 0, stream>>>(t1[i], pk_rb[i][r][1],
                                                  rb_b2[i][r], t2[i], LR[i], 27);
      // out = lrelu(norm(conv2) + residual, 0.2) written back into volB
      kcdna5_inorm<<<dim3(B * 64), dim3(256), 0, stream>>>(
          t2[i], volB[i], volB[i], res3, 0.2f);
    }
  }

  // ---- 6. fused trilinear sampling + FC1(ReLU) + FC2 ----
  kcdna5_mlp<<<dim3((unsigned)((B * (long)N) / 16)), dim3(256), 0, stream>>>(
      qpts, volB[0], volB[1], volB[2], pk_fc1, fc1b, pk_fc2, fc2b, out);

  (void)in_sizes; (void)n_in; (void)out_size; (void)ws_size;
}
